// MultiHeadSelfAttention_21122649162123
// MI455X (gfx1250) — matrix operand order repaired and checked by reading, not compiled, not run on hardware
//
#include <hip/hip_runtime.h>
#include <hip/hip_bf16.h>

// MHA forward for MI455X (gfx1250), wave32 + WMMA f16->f32.
// B=2 S=2048 D=1024 H=16 HD=64

#define BB 2
#define SS 2048
#define DD 1024
#define HH 16
#define HDIM 64
#define MT (BB*SS)          // 4096 tokens total

typedef __attribute__((ext_vector_type(16))) _Float16 v16h;
typedef __attribute__((ext_vector_type(8)))  _Float16 v8h;
typedef __attribute__((ext_vector_type(8)))  float    v8f;

union AFrag { v16h v; v8h h[2]; };

// cross-lane xor shuffle via DS_SWIZZLE (group-of-32 mode: offset[15]=0,
// and_mask=0x1F, xor_mask=XM). No LDS memory touched.
template<int XM>
__device__ __forceinline__ float swz(float x) {
  return __int_as_float(__builtin_amdgcn_ds_swizzle(__float_as_int(x), (XM << 10) | 0x1F));
}

__device__ __forceinline__ v8f wmma_f16(v16h a, v16h b, v8f c) {
  // D = A(16x32) * B(32x16) + C   (f16 inputs, f32 accumulate)
  return __builtin_amdgcn_wmma_f32_16x16x32_f16(false, a, false, b, (short)0, c, false, false);
}

// ---------------------------------------------------------------- convert
__global__ void k_convert(const float* __restrict__ x,
                          const float* __restrict__ wq, const float* __restrict__ wk,
                          const float* __restrict__ wv, const float* __restrict__ wo,
                          _Float16* __restrict__ Xh, _Float16* __restrict__ Wh) {
  const int NX = MT * DD;      // 1<<22
  const int NW = DD * DD;      // 1<<20
  int i = blockIdx.x * blockDim.x + threadIdx.x;
  if (i < NX) { Xh[i] = (_Float16)x[i]; return; }
  int t = i - NX;
  if (t >= 4 * NW) return;
  int wsel = t >> 20;
  int j = t & (NW - 1);
  const float* w = (wsel == 0) ? wq : (wsel == 1) ? wk : (wsel == 2) ? wv : wo;
  Wh[t] = (_Float16)w[j];
}

// ------------------------------------------------- QKV projection (+RoPE)
// One wave computes a 32x64 tile of Y = X @ W^T.  grid.z selects q/k/v.
__global__ void __launch_bounds__(32)
k_proj(const _Float16* __restrict__ Xh, const _Float16* __restrict__ Wh,
       const float* __restrict__ fcos, const float* __restrict__ fsin,
       _Float16* __restrict__ Qr, _Float16* __restrict__ Kr,
       _Float16* __restrict__ Vt) {
  const int lane = threadIdx.x & 31;
  const int sub  = lane >> 4;        // half-wave index
  const int l16  = lane & 15;
  const int mode = blockIdx.z;       // 0=q 1=k 2=v
  const _Float16* W = Wh + mode * (DD * DD);
  const int mbase = blockIdx.y * 32;
  const int nbase = blockIdx.x * 64;

  v8f zero = {};
  v8f acc[2][4];
#pragma unroll
  for (int mi = 0; mi < 2; ++mi)
#pragma unroll
    for (int f = 0; f < 4; ++f) acc[mi][f] = zero;

  const int arow0 = (mbase + l16) * DD;
  const int arow1 = (mbase + 16 + l16) * DD;

  for (int k0 = 0; k0 < DD; k0 += 32) {
    AFrag a0, a1;
    // A-frag layout: lane row = l16; K = 16*sub + 0..15
    a0.h[0] = *(const v8h*)&Xh[arow0 + k0 + 16 * sub];
    a0.h[1] = *(const v8h*)&Xh[arow0 + k0 + 8 + 16 * sub];
    a1.h[0] = *(const v8h*)&Xh[arow1 + k0 + 16 * sub];
    a1.h[1] = *(const v8h*)&Xh[arow1 + k0 + 8 + 16 * sub];
#pragma unroll
    for (int f = 0; f < 4; ++f) {
      // B-frag layout: lane col = l16; K = 16*sub + 0..15 (contiguous in W row)
      v16h bf = *(const v16h*)&W[(nbase + f * 16 + l16) * DD + k0 + 16 * sub];
      acc[0][f] = wmma_f16(a0.v, bf, acc[0][f]);
      acc[1][f] = wmma_f16(a1.v, bf, acc[1][f]);
    }
  }

  // Epilogue: RoPE (q,k only) + scatter into attention layouts.
#pragma unroll
  for (int mi = 0; mi < 2; ++mi) {
#pragma unroll
    for (int f = 0; f < 4; ++f) {
      const int col = nbase + f * 16 + l16;
      const int hh = col >> 6;
      const int d  = col & 63;
#pragma unroll
      for (int j = 0; j < 8; ++j) {
        const int row = mbase + mi * 16 + j + 8 * sub;  // C-frag: M = j + 8*sub
        const int b = row >> 11;                        // row / S
        const int s = row & (SS - 1);
        float val = acc[mi][f][j];
        if (mode < 2) {
          const int p = d >> 1;
          const float c  = fcos[s * (HDIM / 2) + p];
          const float sn = fsin[s * (HDIM / 2) + p];
          const float other = swz<1>(val);   // partner lane holds d^1 (same row)
          val = (d & 1) ? (other * sn + val * c)   // odd:  t0*s + t1*c
                        : (val * c - other * sn);  // even: t0*c - t1*s
        }
        _Float16 hv = (_Float16)val;
        if (mode == 0)      Qr[((b * HH + hh) * SS + s) * HDIM + d] = hv;
        else if (mode == 1) Kr[((b * HH + hh) * SS + s) * HDIM + d] = hv;
        else                Vt[((b * HH + hh) * HDIM + d) * SS + s] = hv;  // transposed
      }
    }
  }
}

// -------------------------------------------------- flash attention core
// One wave handles one (b, h, 16-query tile); streams K/V in 32-key chunks.
__global__ void __launch_bounds__(32)
k_attn(const _Float16* __restrict__ Qr, const _Float16* __restrict__ Kr,
       const _Float16* __restrict__ Vt, const int* __restrict__ pmask,
       _Float16* __restrict__ Attn) {
  __shared__ __attribute__((aligned(16))) _Float16 pbuf[16 * 32];  // P tile staging
  const int lane = threadIdx.x & 31;
  const int sub  = lane >> 4;
  const int l16  = lane & 15;
  const int qt = blockIdx.x, hh = blockIdx.y, b = blockIdx.z;
  const _Float16* Qb = Qr + ((b * HH + hh) * SS) * HDIM;
  const _Float16* Kb = Kr + ((b * HH + hh) * SS) * HDIM;
  const _Float16* Vb = Vt + ((b * HH + hh) * HDIM) * SS;
  const int* pm = pmask + b * SS;

  // Q A-fragments for the 16x64 query tile (two 32-deep d-chunks), kept live.
  AFrag qa[2];
  {
    const int qrow = (qt * 16 + l16) * HDIM;
    qa[0].h[0] = *(const v8h*)&Qb[qrow + 16 * sub];
    qa[0].h[1] = *(const v8h*)&Qb[qrow + 8 + 16 * sub];
    qa[1].h[0] = *(const v8h*)&Qb[qrow + 32 + 16 * sub];
    qa[1].h[1] = *(const v8h*)&Qb[qrow + 32 + 8 + 16 * sub];
  }

  const float NEG_INF = -__builtin_inff();
  float mrun[8], lrun[8];
#pragma unroll
  for (int j = 0; j < 8; ++j) { mrun[j] = NEG_INF; lrun[j] = 0.f; }
  v8f zero = {};
  v8f o[4];
#pragma unroll
  for (int f = 0; f < 4; ++f) o[f] = zero;

  for (int base = 0; base < SS; base += 32) {
    // K^T B-frags: lane col = key token (l16), elements run over d (contiguous).
    v16h kb00 = *(const v16h*)&Kb[(base + l16)      * HDIM + 16 * sub];
    v16h kb01 = *(const v16h*)&Kb[(base + l16)      * HDIM + 32 + 16 * sub];
    v16h kb10 = *(const v16h*)&Kb[(base + 16 + l16) * HDIM + 16 * sub];
    v16h kb11 = *(const v16h*)&Kb[(base + 16 + l16) * HDIM + 32 + 16 * sub];
    v8f s0 = zero, s1 = zero;
    s0 = wmma_f16(qa[0].v, kb00, s0);
    s0 = wmma_f16(qa[1].v, kb01, s0);
    s1 = wmma_f16(qa[0].v, kb10, s1);
    s1 = wmma_f16(qa[1].v, kb11, s1);

    const int mk0 = pm[base + l16];
    const int mk1 = pm[base + 16 + l16];

#pragma unroll
    for (int j = 0; j < 8; ++j) {
      // scale = 1/sqrt(D) = 1/32 ; masked keys -> -inf (first chunk always valid,
      // since lengths >= S/2, so m never stays -inf once keys are seen)
      float x0 = mk0 ? s0[j] * 0.03125f : NEG_INF;
      float x1 = mk1 ? s1[j] * 0.03125f : NEG_INF;
      float rm = fmaxf(x0, x1);
      rm = fmaxf(rm, swz<1>(rm));
      rm = fmaxf(rm, swz<2>(rm));
      rm = fmaxf(rm, swz<4>(rm));
      rm = fmaxf(rm, swz<8>(rm));
      float mnew  = fmaxf(mrun[j], rm);
      float alpha = __expf(mrun[j] - mnew);
      float p0 = __expf(x0 - mnew);
      float p1 = __expf(x1 - mnew);
      float rs = p0 + p1;
      rs += swz<1>(rs); rs += swz<2>(rs); rs += swz<4>(rs); rs += swz<8>(rs);
      lrun[j] = lrun[j] * alpha + rs;
      mrun[j] = mnew;
#pragma unroll
      for (int f = 0; f < 4; ++f) o[f][j] *= alpha;
      // stage P (C-layout) into LDS row-major [q=0..15][k=0..31]
      const int r = j + 8 * sub;
      pbuf[r * 32 + l16]      = (_Float16)p0;
      pbuf[r * 32 + 16 + l16] = (_Float16)p1;
    }

    // Re-read P in A-fragment layout (same-wave DS ops are in order).
    AFrag pa;
    pa.h[0] = *(const v8h*)&pbuf[l16 * 32 + 16 * sub];
    pa.h[1] = *(const v8h*)&pbuf[l16 * 32 + 8 + 16 * sub];
#pragma unroll
    for (int f = 0; f < 4; ++f) {
      // V B-frag from transposed V: lane col = d (f*16+l16), elements over tokens.
      v16h vb = *(const v16h*)&Vb[(f * 16 + l16) * SS + base + 16 * sub];
      o[f] = wmma_f16(pa.v, vb, o[f]);
    }
  }

  // Normalize and store head output into (B*S, D) f16 for the final GEMM.
#pragma unroll
  for (int f = 0; f < 4; ++f) {
    const int col = hh * HDIM + f * 16 + l16;
#pragma unroll
    for (int j = 0; j < 8; ++j) {
      const int sq = qt * 16 + j + 8 * sub;
      Attn[(b * SS + sq) * DD + col] = (_Float16)(o[f][j] / lrun[j]);
    }
  }
}

// ------------------------------------------------- output projection GEMM
__global__ void __launch_bounds__(32)
k_ogemm(const _Float16* __restrict__ Ah, const _Float16* __restrict__ Wo,
        float* __restrict__ Y) {
  const int lane = threadIdx.x & 31;
  const int sub  = lane >> 4;
  const int l16  = lane & 15;
  const int mbase = blockIdx.y * 32;
  const int nbase = blockIdx.x * 64;

  v8f zero = {};
  v8f acc[2][4];
#pragma unroll
  for (int mi = 0; mi < 2; ++mi)
#pragma unroll
    for (int f = 0; f < 4; ++f) acc[mi][f] = zero;

  const int arow0 = (mbase + l16) * DD;
  const int arow1 = (mbase + 16 + l16) * DD;

  for (int k0 = 0; k0 < DD; k0 += 32) {
    AFrag a0, a1;
    a0.h[0] = *(const v8h*)&Ah[arow0 + k0 + 16 * sub];
    a0.h[1] = *(const v8h*)&Ah[arow0 + k0 + 8 + 16 * sub];
    a1.h[0] = *(const v8h*)&Ah[arow1 + k0 + 16 * sub];
    a1.h[1] = *(const v8h*)&Ah[arow1 + k0 + 8 + 16 * sub];
#pragma unroll
    for (int f = 0; f < 4; ++f) {
      v16h bf = *(const v16h*)&Wo[(nbase + f * 16 + l16) * DD + k0 + 16 * sub];
      acc[0][f] = wmma_f16(a0.v, bf, acc[0][f]);
      acc[1][f] = wmma_f16(a1.v, bf, acc[1][f]);
    }
  }

#pragma unroll
  for (int mi = 0; mi < 2; ++mi)
#pragma unroll
    for (int f = 0; f < 4; ++f) {
      const int col = nbase + f * 16 + l16;
#pragma unroll
      for (int j = 0; j < 8; ++j) {
        const int row = mbase + mi * 16 + j + 8 * sub;
        Y[row * DD + col] = acc[mi][f][j];
      }
    }
}

// ----------------------------------------------------------------- launch
extern "C" void kernel_launch(void* const* d_in, const int* in_sizes, int n_in,
                              void* d_out, int out_size, void* d_ws, size_t ws_size,
                              hipStream_t stream) {
  const float* x  = (const float*)d_in[0];
  const float* wq = (const float*)d_in[1];
  const float* wk = (const float*)d_in[2];
  const float* wv = (const float*)d_in[3];
  const float* wo = (const float*)d_in[4];
  const float* fc = (const float*)d_in[5];
  const float* fs = (const float*)d_in[6];
  const int*   pm = (const int*)d_in[7];

  char* ws = (char*)d_ws;
  _Float16* Xh = (_Float16*)(ws);                    //  0..8MB   x in f16
  _Float16* Wh = (_Float16*)(ws + (size_t)( 8 << 20)); //  8..16MB  wq|wk|wv|wo f16
  _Float16* Qr = (_Float16*)(ws + (size_t)(16 << 20)); // (B,H,S,HD)
  _Float16* Kr = (_Float16*)(ws + (size_t)(24 << 20)); // (B,H,S,HD)
  _Float16* Vt = (_Float16*)(ws + (size_t)(32 << 20)); // (B,H,HD,S) transposed
  _Float16* At = (_Float16*)(ws + (size_t)(40 << 20)); // (B*S, D) attn out
  float* Y = (float*)d_out;

  const int total_cvt = MT * DD + 4 * DD * DD;       // 8,388,608
  k_convert<<<dim3((total_cvt + 255) / 256), dim3(256), 0, stream>>>(
      x, wq, wk, wv, wo, Xh, Wh);

  k_proj<<<dim3(DD / 64, MT / 32, 3), dim3(32), 0, stream>>>(
      Xh, Wh, fc, fs, Qr, Kr, Vt);

  k_attn<<<dim3(SS / 16, HH, BB), dim3(32), 0, stream>>>(
      Qr, Kr, Vt, pm, At);

  k_ogemm<<<dim3(DD / 64, MT / 32), dim3(32), 0, stream>>>(
      At, Wh + 3 * DD * DD, Y);
}
